// CNNMEAE_71064528879932
// MI455X (gfx1250) — matrix operand order repaired and checked by reading, not compiled, not run on hardware
//
#include <hip/hip_runtime.h>
#include <hip/hip_bf16.h>

// ---------------------------------------------------------------------------
// Types for CDNA5 WMMA (wave32, 16x16x32 bf16 -> f32) and TDM descriptors
// ---------------------------------------------------------------------------
typedef __attribute__((ext_vector_type(16))) __bf16 v16bf;
typedef __attribute__((ext_vector_type(8)))  float  v8f;
typedef __attribute__((ext_vector_type(4)))  unsigned int v4u;
typedef __attribute__((ext_vector_type(8)))  int    v8i;
typedef __attribute__((ext_vector_type(4)))  int    v4i;

union FragBF16 { v16bf v; uint4 q[2]; };

__device__ __forceinline__ unsigned short f32_to_bf16(float f) {
    union { float f; unsigned u; } v; v.f = f;
    unsigned r = v.u + 0x7FFFu + ((v.u >> 16) & 1u);   // round-to-nearest-even
    return (unsigned short)(r >> 16);
}

#define LRELU(v) ((v) >= 0.f ? (v) : 0.01f * (v))

// ---------------------------------------------------------------------------
// Conv1d (NCH, OIH weights), direct form
// ---------------------------------------------------------------------------
__global__ void conv1d_k(const float* __restrict__ in, const float* __restrict__ w,
                         const float* __restrict__ bias, float* __restrict__ out,
                         int Bn, int Ci, int Li, int Co, int Lo, int s, int p) {
    int idx = blockIdx.x * blockDim.x + threadIdx.x;
    int total = Bn * Co * Lo;
    if (idx >= total) return;
    int ol = idx % Lo; int t = idx / Lo; int oc = t % Co; int b = t / Co;
    float acc = bias[oc];
    int il0 = ol * s - p;
    for (int ic = 0; ic < Ci; ++ic) {
        const float* ip = in + (b * Ci + ic) * Li;
        const float* wp = w + (oc * Ci + ic) * 3;
#pragma unroll
        for (int k = 0; k < 3; ++k) {
            int il = il0 + k;
            if (il >= 0 && il < Li) acc += ip[il] * wp[k];
        }
    }
    out[idx] = acc;
}

// ---------------------------------------------------------------------------
// ConvTranspose1d (torch semantics, gather form), optional fused sigmoid
// ---------------------------------------------------------------------------
__global__ void deconv1d_k(const float* __restrict__ in, const float* __restrict__ w,
                           const float* __restrict__ bias, float* __restrict__ out,
                           int Bn, int Ci, int Li, int Co, int Lo, int s, int p, int do_sig) {
    int idx = blockIdx.x * blockDim.x + threadIdx.x;
    int total = Bn * Co * Lo;
    if (idx >= total) return;
    int ol = idx % Lo; int t = idx / Lo; int oc = t % Co; int b = t / Co;
    float acc = bias[oc];
    for (int ic = 0; ic < Ci; ++ic) {
        const float* ip = in + (b * Ci + ic) * Li;
        const float* wp = w + (ic * Co + oc) * 3;
#pragma unroll
        for (int k = 0; k < 3; ++k) {
            int tt = ol + p - k;
            if (tt >= 0 && (tt % s) == 0) {
                int il = tt / s;
                if (il < Li) acc += ip[il] * wp[k];
            }
        }
    }
    if (do_sig) acc = 1.f / (1.f + expf(-acc));
    out[idx] = acc;
}

// ---------------------------------------------------------------------------
// Training-mode BN stats: one block per channel, reduce over (N, L)
// ---------------------------------------------------------------------------
__global__ void bn_stats_k(const float* __restrict__ x, float* __restrict__ mean,
                           float* __restrict__ istd, int Bn, int C, int L) {
    int c = blockIdx.x;
    int n = Bn * L;
    float s = 0.f, s2 = 0.f;
    for (int i = threadIdx.x; i < n; i += blockDim.x) {
        int b = i / L, l = i - b * L;
        float v = x[(b * C + c) * L + l];
        s += v; s2 += v * v;
    }
    __shared__ float sh[256], sh2[256];
    sh[threadIdx.x] = s; sh2[threadIdx.x] = s2;
    __syncthreads();
    for (int o = blockDim.x >> 1; o > 0; o >>= 1) {
        if ((int)threadIdx.x < o) { sh[threadIdx.x] += sh[threadIdx.x + o]; sh2[threadIdx.x] += sh2[threadIdx.x + o]; }
        __syncthreads();
    }
    if (threadIdx.x == 0) {
        float m = sh[0] / (float)n;
        float v = sh2[0] / (float)n - m * m;
        mean[c] = m;
        istd[c] = rsqrtf(v + 1e-5f);
    }
}

__global__ void bn_apply_lrelu_k(float* __restrict__ x, const float* __restrict__ mean,
                                 const float* __restrict__ istd, const float* __restrict__ g,
                                 const float* __restrict__ be, int Bn, int C, int L) {
    int idx = blockIdx.x * blockDim.x + threadIdx.x;
    int total = Bn * C * L;
    if (idx >= total) return;
    int c = (idx / L) % C;
    float v = (x[idx] - mean[c]) * istd[c] * g[c] + be[c];
    x[idx] = LRELU(v);
}

// ---------------------------------------------------------------------------
// z row-normalize -> bf16 padded (256 x 992)
// ---------------------------------------------------------------------------
__global__ void zn_k(const float* __restrict__ z, unsigned short* __restrict__ zn,
                     int F, int KP) {
    int b = blockIdx.x;
    const float* zr = z + b * F;
    float s = 0.f;
    for (int i = threadIdx.x; i < F; i += blockDim.x) { float v = zr[i]; s += v * v; }
    __shared__ float sh[256];
    sh[threadIdx.x] = s; __syncthreads();
    for (int o = blockDim.x >> 1; o > 0; o >>= 1) {
        if ((int)threadIdx.x < o) sh[threadIdx.x] += sh[threadIdx.x + o];
        __syncthreads();
    }
    float inv = 1.f / fmaxf(sqrtf(sh[0]), 1e-8f);
    for (int i = threadIdx.x; i < KP; i += blockDim.x) {
        float v = (i < F) ? zr[i] * inv : 0.f;
        zn[b * KP + i] = f32_to_bf16(v);
    }
}

// ---------------------------------------------------------------------------
// mem: inverse row norms + bf16 copy padded (20000 x 992)
// ---------------------------------------------------------------------------
__global__ void memprep_k(const float* __restrict__ mem, unsigned short* __restrict__ mb,
                          float* __restrict__ invn, int F, int KP) {
    int m = blockIdx.x;
    const float* mr = mem + (size_t)m * F;
    float s = 0.f;
    for (int i = threadIdx.x; i < F; i += blockDim.x) { float v = mr[i]; s += v * v; }
    __shared__ float sh[256];
    sh[threadIdx.x] = s; __syncthreads();
    for (int o = blockDim.x >> 1; o > 0; o >>= 1) {
        if ((int)threadIdx.x < o) sh[threadIdx.x] += sh[threadIdx.x + o];
        __syncthreads();
    }
    if (threadIdx.x == 0) invn[m] = 1.f / fmaxf(sqrtf(sh[0]), 1e-8f);
    for (int i = threadIdx.x; i < KP; i += blockDim.x)
        mb[(size_t)m * KP + i] = f32_to_bf16(i < F ? mr[i] : 0.f);
}

// feat-major transposed bf16 copy: memT[f*M + m] = mem[m*F + f], f padded to KP
__global__ void memT_k(const float* __restrict__ mem, unsigned short* __restrict__ mt,
                       int M, int F, int KP) {
    size_t idx = (size_t)blockIdx.x * blockDim.x + threadIdx.x;
    size_t total = (size_t)KP * M;
    if (idx >= total) return;
    int m = (int)(idx % M);
    int f = (int)(idx / M);
    mt[idx] = f32_to_bf16(f < F ? mem[(size_t)m * F + f] : 0.f);
}

// ---------------------------------------------------------------------------
// GEMM1: logit[b,n] = (zn[b,:] . mem[n,:]) * invnorm[n]
// A: 256 x KP (row-major bf16) -- staged per-block into LDS by the Tensor
// Data Mover (tensor_load_to_lds, TENSORcnt). Bm: 20000 x KP streamed from
// global (L2-resident). One wave per 16x16 tile, 8 n-tiles per block.
// ---------------------------------------------------------------------------
__global__ void __launch_bounds__(256)
gemm1_wmma_k(const unsigned short* __restrict__ A,
             const unsigned short* __restrict__ Bm,
             const float* __restrict__ invn,
             float* __restrict__ out,
             int Mrows, int Ncols, int KP, int nchunk) {
    __shared__ __align__(16) unsigned short shA[16 * 992];   // 31,744 B

    int mt    = blockIdx.x / nchunk;
    int chunk = blockIdx.x - mt * nchunk;
    int wid   = threadIdx.x >> 5;
    int lane  = threadIdx.x & 31;
    int ntiles = Ncols >> 4;
    int nt = chunk * 8 + wid;
    int m0 = mt << 4;

    // ---- stage A tile (16 x KP bf16) into LDS via TDM (wave 0 only) ----
    if (wid == 0) {
        unsigned lds_addr = (unsigned)(size_t)(void*)shA;   // low 32 bits = LDS offset
        unsigned long long ga = (unsigned long long)(size_t)(const void*)(A + (size_t)m0 * KP);
        // D# group 0: count=1 | lds_addr | global_addr | type=2
        v4u g0;
        g0.x = 1u;                                  // count=1, is_restore=0, gather=0
        g0.y = lds_addr;                            // LDS byte address
        g0.z = (unsigned)(ga & 0xFFFFFFFFu);        // global_addr[31:0]
        g0.w = (unsigned)(ga >> 32) | 0x80000000u;  // global_addr[56:32] | type=2<<30
        // D# group 1: mask=0, data_size=2B, 2-D tile KP x 16, stride KP
        v8i g1;
        g1[0] = (int)(1u << 16);                    // data_size=1 (2 bytes)
        g1[1] = (int)((unsigned)KP << 16);          // tensor_dim0[15:0] (=KP)
        g1[2] = (int)(16u << 16);                   // tensor_dim0[31:16]=0 | tensor_dim1=16
        g1[3] = (int)((unsigned)KP << 16);          // tensor_dim1[31:16]=0 | tile_dim0=KP
        g1[4] = 16;                                 // tile_dim1=16 | tile_dim2=0
        g1[5] = KP;                                 // tensor_dim0_stride[31:0]
        g1[6] = 0;                                  // stride hi | dim1_stride lo
        g1[7] = 0;
        v4i gz4 = {0, 0, 0, 0};                     // groups 2/3 unused (2-D tensor)
        v8i gz8 = {0, 0, 0, 0, 0, 0, 0, 0};        // extra group (clang-23 6-arg form)
        __builtin_amdgcn_tensor_load_to_lds(g0, g1, gz4, gz4, gz8, 0);
        __builtin_amdgcn_s_wait_tensorcnt(0);       // TENSORcnt -> 0: tile resident
    }
    __syncthreads();

    if (nt < ntiles && mt < (Mrows >> 4)) {
        int n0 = nt << 4;
        int lrow = lane & 15;
        int hi = lane >> 4;                 // 0 or 1
        const unsigned short* arow = shA + lrow * KP + (hi ? 8 : 0);          // LDS
        const unsigned short* brow = Bm + (size_t)(n0 + lrow) * KP + (hi ? 8 : 0);
        v8f acc = {};
        for (int kb = 0; kb < KP; kb += 32) {
            FragBF16 af, bf;
            af.q[0] = *(const uint4*)(arow + kb);        // ds_load_b128
            af.q[1] = *(const uint4*)(arow + kb + 16);
            bf.q[0] = *(const uint4*)(brow + kb);        // global_load_b128
            bf.q[1] = *(const uint4*)(brow + kb + 16);
            if (kb + 32 < KP) __builtin_prefetch((const void*)(brow + kb + 32), 0, 1);
            acc = __builtin_amdgcn_wmma_f32_16x16x32_bf16(false, af.v, false, bf.v,
                                                          (short)0, acc, false, false);
        }
        int col = n0 + lrow;
        float sc = invn[col];
        int rbase = m0 + (hi ? 8 : 0);
#pragma unroll
        for (int r = 0; r < 8; ++r)
            out[(size_t)(rbase + r) * Ncols + col] = acc[r] * sc;
    }
}

// ---------------------------------------------------------------------------
// Softmax + hard shrinkage + L1 normalize per row (in-place), emit bf16 copy
// ---------------------------------------------------------------------------
__global__ void softmax_shrink_k(float* __restrict__ w, unsigned short* __restrict__ wb, int Mn) {
    int b = blockIdx.x, tid = threadIdx.x;
    float* row = w + (size_t)b * Mn;
    __shared__ float sh[256];
    // max
    float mx = -3.4e38f;
    for (int i = tid; i < Mn; i += 256) mx = fmaxf(mx, row[i]);
    sh[tid] = mx; __syncthreads();
    for (int o = 128; o > 0; o >>= 1) { if (tid < o) sh[tid] = fmaxf(sh[tid], sh[tid + o]); __syncthreads(); }
    mx = sh[0]; __syncthreads();
    // exp + sum
    float s = 0.f;
    for (int i = tid; i < Mn; i += 256) { float e = expf(row[i] - mx); row[i] = e; s += e; }
    sh[tid] = s; __syncthreads();
    for (int o = 128; o > 0; o >>= 1) { if (tid < o) sh[tid] += sh[tid + o]; __syncthreads(); }
    float inv = 1.f / sh[0]; __syncthreads();
    // shrinkage + sum
    const float TH = 1.f / 20000.f;
    float s2 = 0.f;
    for (int i = tid; i < Mn; i += 256) {
        float wv = row[i] * inv;
        float d = wv - TH;
        float sv = fmaxf(d, 0.f) * wv / (fabsf(d) + 1e-15f);
        row[i] = sv; s2 += sv;
    }
    sh[tid] = s2; __syncthreads();
    for (int o = 128; o > 0; o >>= 1) { if (tid < o) sh[tid] += sh[tid + o]; __syncthreads(); }
    float inv2 = 1.f / sh[0];
    for (int i = tid; i < Mn; i += 256) {
        float v = row[i] * inv2;
        row[i] = v;
        wb[(size_t)b * Mn + i] = f32_to_bf16(v);
    }
}

// ---------------------------------------------------------------------------
// GEMM2: z_hat[b,f] = sum_m w[b,m] * mem[m,f]
// A: 256 x K (w_bf16, row-major), Bt: feat-major memT (NP x K), K = 20000
// ---------------------------------------------------------------------------
__global__ void gemm2_wmma_k(const unsigned short* __restrict__ A,
                             const unsigned short* __restrict__ Bt,
                             float* __restrict__ zhat,
                             int Mrows, int NP, int K, int F) {
    int wave = (blockIdx.x * blockDim.x + threadIdx.x) >> 5;
    int lane = threadIdx.x & 31;
    int ntiles = NP >> 4;
    int wtotal = (Mrows >> 4) * ntiles;
    if (wave >= wtotal) return;
    int mt = wave / ntiles, nt = wave - mt * ntiles;
    int m0 = mt << 4, n0 = nt << 4;
    int lrow = lane & 15;
    int hi = lane >> 4;
    const unsigned short* arow = A  + (size_t)(m0 + lrow) * K + (hi ? 8 : 0);
    const unsigned short* brow = Bt + (size_t)(n0 + lrow) * K + (hi ? 8 : 0);
    v8f acc = {};
    for (int kb = 0; kb < K; kb += 32) {
        FragBF16 af, bf;
        af.q[0] = *(const uint4*)(arow + kb);
        af.q[1] = *(const uint4*)(arow + kb + 16);
        bf.q[0] = *(const uint4*)(brow + kb);
        bf.q[1] = *(const uint4*)(brow + kb + 16);
        if (kb + 32 < K) {
            __builtin_prefetch((const void*)(arow + kb + 32), 0, 1);
            __builtin_prefetch((const void*)(brow + kb + 32), 0, 1);
        }
        acc = __builtin_amdgcn_wmma_f32_16x16x32_bf16(false, af.v, false, bf.v,
                                                      (short)0, acc, false, false);
    }
    int col = n0 + lrow;
    if (col < F) {
        int rbase = m0 + (hi ? 8 : 0);
#pragma unroll
        for (int r = 0; r < 8; ++r)
            zhat[(size_t)(rbase + r) * F + col] = acc[r];
    }
}

// ---------------------------------------------------------------------------
// Host-side orchestration
// ---------------------------------------------------------------------------
extern "C" void kernel_launch(void* const* d_in, const int* in_sizes, int n_in,
                              void* d_out, int out_size, void* d_ws, size_t ws_size,
                              hipStream_t stream) {
    const float* x    = (const float*)d_in[0];
    const float* ew1  = (const float*)d_in[1];  const float* eb1  = (const float*)d_in[2];
    const float* eg1  = (const float*)d_in[3];  const float* ebe1 = (const float*)d_in[4];
    const float* ew2  = (const float*)d_in[5];  const float* eb2  = (const float*)d_in[6];
    const float* eg2  = (const float*)d_in[7];  const float* ebe2 = (const float*)d_in[8];
    const float* ew3  = (const float*)d_in[9];  const float* eb3  = (const float*)d_in[10];
    const float* eg3  = (const float*)d_in[11]; const float* ebe3 = (const float*)d_in[12];
    const float* ew4  = (const float*)d_in[13]; const float* eb4  = (const float*)d_in[14];
    const float* eg4  = (const float*)d_in[15]; const float* ebe4 = (const float*)d_in[16];
    const float* dw1  = (const float*)d_in[17]; const float* db1  = (const float*)d_in[18];
    const float* dg1  = (const float*)d_in[19]; const float* dbe1 = (const float*)d_in[20];
    const float* dw2  = (const float*)d_in[21]; const float* db2  = (const float*)d_in[22];
    const float* dg2  = (const float*)d_in[23]; const float* dbe2 = (const float*)d_in[24];
    const float* dw3  = (const float*)d_in[25]; const float* db3  = (const float*)d_in[26];
    const float* dg3  = (const float*)d_in[27]; const float* dbe3 = (const float*)d_in[28];
    const float* dw4  = (const float*)d_in[29]; const float* db4  = (const float*)d_in[30];
    const float* mem  = (const float*)d_in[31];

    const int Bn = 256, F = 984, KP = 992, Mn = 20000;

    float* out_rec = (float*)d_out;            // 256*1000
    float* out_w   = out_rec + 256 * 1000;     // 256*20000
    float* out_z   = out_w + 256 * 20000;      // 256*984

    // workspace carve
    float* buf0     = (float*)d_ws;                       // 256*64*250 = 4,096,000
    float* buf1     = buf0 + 4096000;                     // 4,096,000
    float* stats    = buf1 + 4096000;                     // 128 (mean[64] + istd[64])
    float* inv_norm = stats + 128;                        // 20000
    float* z_hat    = inv_norm + 20000;                   // 256*984
    unsigned short* zn_bf   = (unsigned short*)(z_hat + 256 * 984);  // 256*992
    unsigned short* w_bf    = zn_bf + 256 * 992;                     // 256*20000
    unsigned short* mem_bf  = w_bf + 256 * 20000;                    // 20000*992
    unsigned short* memT_bf = mem_bf + (size_t)Mn * KP;              // 992*20000

    dim3 blk(256);
    auto G = [](long long n) { return dim3((unsigned)((n + 255) / 256)); };

    // ---------------- encoder ----------------
    conv1d_k<<<G(256LL*8*500), blk, 0, stream>>>(x, ew1, eb1, buf0, Bn, 1, 1000, 8, 500, 2, 1);
    bn_stats_k<<<8, blk, 0, stream>>>(buf0, stats, stats + 64, Bn, 8, 500);
    bn_apply_lrelu_k<<<G(256LL*8*500), blk, 0, stream>>>(buf0, stats, stats + 64, eg1, ebe1, Bn, 8, 500);

    conv1d_k<<<G(256LL*64*250), blk, 0, stream>>>(buf0, ew2, eb2, buf1, Bn, 8, 500, 64, 250, 2, 1);
    bn_stats_k<<<64, blk, 0, stream>>>(buf1, stats, stats + 64, Bn, 64, 250);
    bn_apply_lrelu_k<<<G(256LL*64*250), blk, 0, stream>>>(buf1, stats, stats + 64, eg2, ebe2, Bn, 64, 250);

    conv1d_k<<<G(256LL*64*125), blk, 0, stream>>>(buf1, ew3, eb3, buf0, Bn, 64, 250, 64, 125, 2, 1);
    bn_stats_k<<<64, blk, 0, stream>>>(buf0, stats, stats + 64, Bn, 64, 125);
    bn_apply_lrelu_k<<<G(256LL*64*125), blk, 0, stream>>>(buf0, stats, stats + 64, eg3, ebe3, Bn, 64, 125);

    conv1d_k<<<G(256LL*8*123), blk, 0, stream>>>(buf0, ew4, eb4, out_z, Bn, 64, 125, 8, 123, 1, 0);
    bn_stats_k<<<8, blk, 0, stream>>>(out_z, stats, stats + 64, Bn, 8, 123);
    bn_apply_lrelu_k<<<G(256LL*8*123), blk, 0, stream>>>(out_z, stats, stats + 64, eg4, ebe4, Bn, 8, 123);

    // ---------------- memory addressing ----------------
    zn_k<<<Bn, blk, 0, stream>>>(out_z, zn_bf, F, KP);
    memprep_k<<<Mn, blk, 0, stream>>>(mem, mem_bf, inv_norm, F, KP);
    memT_k<<<G((long long)KP * Mn), blk, 0, stream>>>(mem, memT_bf, Mn, F, KP);

    // logits: 16 m-tiles * 157 n-chunks (8 n-tiles each, one per wave)
    const int nchunk = (1250 + 7) / 8;   // 157
    gemm1_wmma_k<<<dim3(16 * nchunk), blk, 0, stream>>>(zn_bf, mem_bf, inv_norm, out_w,
                                                        Bn, Mn, KP, nchunk);

    softmax_shrink_k<<<Bn, blk, 0, stream>>>(out_w, w_bf, Mn);

    // z_hat: 16 m-tiles * 62 n-tiles = 992 waves -> 124 blocks
    gemm2_wmma_k<<<124, blk, 0, stream>>>(w_bf, memT_bf, z_hat, Bn, KP, Mn, F);

    // ---------------- decoder ----------------
    deconv1d_k<<<G(256LL*64*125), blk, 0, stream>>>(z_hat, dw1, db1, buf0, Bn, 8, 123, 64, 125, 1, 0, 0);
    bn_stats_k<<<64, blk, 0, stream>>>(buf0, stats, stats + 64, Bn, 64, 125);
    bn_apply_lrelu_k<<<G(256LL*64*125), blk, 0, stream>>>(buf0, stats, stats + 64, dg1, dbe1, Bn, 64, 125);

    deconv1d_k<<<G(256LL*64*250), blk, 0, stream>>>(buf0, dw2, db2, buf1, Bn, 64, 125, 64, 250, 2, 1, 0);
    bn_stats_k<<<64, blk, 0, stream>>>(buf1, stats, stats + 64, Bn, 64, 250);
    bn_apply_lrelu_k<<<G(256LL*64*250), blk, 0, stream>>>(buf1, stats, stats + 64, dg2, dbe2, Bn, 64, 250);

    deconv1d_k<<<G(256LL*8*500), blk, 0, stream>>>(buf1, dw3, db3, buf0, Bn, 64, 250, 8, 500, 2, 1, 0);
    bn_stats_k<<<8, blk, 0, stream>>>(buf0, stats, stats + 64, Bn, 8, 500);
    bn_apply_lrelu_k<<<G(256LL*8*500), blk, 0, stream>>>(buf0, stats, stats + 64, dg3, dbe3, Bn, 8, 500);

    deconv1d_k<<<G(256LL*1*1000), blk, 0, stream>>>(buf0, dw4, db4, out_rec, Bn, 8, 500, 1, 1000, 2, 1, 1);
}
